// RegionProposalNetwork_52879637348354
// MI455X (gfx1250) — matrix operand order repaired and checked by reading, not compiled, not run on hardware
//
#include <hip/hip_runtime.h>
#include <cstdint>
#include <cstddef>

typedef __attribute__((ext_vector_type(16))) _Float16 v16h;
typedef __attribute__((ext_vector_type(8)))  _Float16 v8h;
typedef __attribute__((ext_vector_type(8)))  float    v8f;
typedef __attribute__((ext_vector_type(4)))  float    f4v;

namespace {
constexpr int BATCH = 4;
constexpr int CCH   = 256;          // channels
constexpr int KT3   = 72;           // 3x3 conv reduction tiles: 9*256/32
constexpr int A0N   = 64 * 64 * 3;  // anchors level0 per batch = 12288
constexpr int A1N   = 32 * 32 * 3;  // anchors level1 per batch = 3072
constexpr int ATOT  = A0N + A1N;    // 15360
constexpr int KTOP  = 4608;         // top-k
constexpr int PKEEP = 1000;
constexpr int NSORT = 16384;        // pow2 >= ATOT
constexpr int SUPW  = (KTOP + 31) / 32;  // 144 suppression words per row
}

__device__ __forceinline__ float clamp256(float v) {
  return fminf(fmaxf(v, 0.f), 256.f);
}

// ---------------------------------------------------------------------------
// Streaming f32 -> f16 conversion of activations (NHWC order preserved).
// Each thread converts 4 elements (one float4 load, one 8B store).
__global__ __launch_bounds__(256) void cvt_f16(const float* __restrict__ src,
                                               _Float16* __restrict__ dst) {
  const size_t i = ((size_t)blockIdx.x * 256 + threadIdx.x) * 4;
  f4v f = *(const f4v*)(src + i);
  _Float16 h[4];
#pragma unroll
  for (int j = 0; j < 4; ++j) h[j] = (_Float16)f[j];
  *(unsigned long long*)(dst + i) = *(const unsigned long long*)h;
}

// ---------------------------------------------------------------------------
// Pack 3x3 conv weights (HWIO f32, K = tap*256+ci major, N = co) into the
// per-lane WMMA B-fragment layout: [kt][nt][lane][16 halves],
// lane<16 -> K = kt*32 + h, lane>=16 -> K = kt*32 + 16 + h, N = nt*16+lane%16.
__global__ __launch_bounds__(256) void pack_w3(const float* __restrict__ w,
                                               _Float16* __restrict__ out) {
  const int gid  = blockIdx.x * 256 + threadIdx.x;      // 72*16*32*16 = 589824
  const int h    = gid & 15;
  const int lane = (gid >> 4) & 31;
  const int nt   = (gid >> 9) & 15;
  const int kt   = gid >> 13;
  const int n = nt * 16 + (lane & 15);
  const int k = kt * 32 + ((lane >> 4) << 4) + h;
  out[gid] = (_Float16)w[(size_t)k * CCH + n];
}

// Pack 1x1 head weights: combined B (K=256, N=16): n<12 bbox, 12..14 conf, 15 pad.
__global__ __launch_bounds__(256) void pack_whead(const float* __restrict__ bw,
                                                  const float* __restrict__ cw,
                                                  _Float16* __restrict__ out) {
  const int gid  = blockIdx.x * 256 + threadIdx.x;      // 8*32*16 = 4096
  const int h    = gid & 15;
  const int lane = (gid >> 4) & 31;
  const int kt   = gid >> 9;
  const int n = lane & 15;
  const int k = kt * 32 + ((lane >> 4) << 4) + h;
  float v = 0.f;
  if (n < 12)      v = bw[(size_t)k * 12 + n];
  else if (n < 15) v = cw[(size_t)k * 3 + (n - 12)];
  out[gid] = (_Float16)v;
}

// ---------------------------------------------------------------------------
// Implicit-GEMM 3x3 conv + bias + ReLU, NHWC f16 in -> f16 feat out.
// One wave computes a 16M x 64N tile (4 accumulators). M = B*H*W, N = 256.
// A-fragment layout: lane<16 -> K in [0..15], lane>=16 -> K in [16..31].
template <int H>
__global__ __launch_bounds__(128)
void conv3_wmma(const _Float16* __restrict__ xh, const float* __restrict__ bias,
                const _Float16* __restrict__ bp, _Float16* __restrict__ feat) {
  constexpr int W = H;
  const int lane  = threadIdx.x & 31;
  const int gw    = blockIdx.x * 4 + (threadIdx.x >> 5);
  const int mtile = gw >> 2;
  const int nblk  = gw & 3;
  const int hi    = lane >> 4;          // 0 or 1 (half-wave)
  const int m     = mtile * 16 + (lane & 15);
  const int hw    = H * W;
  const int b     = m / hw;
  const int p     = m % hw;
  const int y     = p / W;
  const int xq    = p % W;

  v8f acc[4] = {};
  const int nt0 = nblk * 4;
  const _Float16* const xbase = xh + (size_t)b * hw * CCH;
  for (int kt = 0; kt < KT3; ++kt) {
    const int tap   = kt >> 3;                      // 256/32 == 8 k-tiles per tap
    const int iy    = y + tap / 3 - 1;
    const int ix    = xq + tap % 3 - 1;
    const int cbase = ((kt & 7) << 5) + (hi << 4);  // one 16-wide group: cbase
    v16h a;
    if ((unsigned)iy < (unsigned)H && (unsigned)ix < (unsigned)W) {
      const _Float16* src = xbase + ((size_t)iy * W + ix) * CCH + cbase;
      a = *(const v16h*)(src);
    } else {
#pragma unroll
      for (int j = 0; j < 16; ++j) a[j] = (_Float16)0.f;
    }

    // Keep the next tap's activation row warm in the WGP path.
    if ((kt & 7) == 0 && tap < 8) {
      const int ny = y + (tap + 1) / 3 - 1;
      const int nx = xq + (tap + 1) % 3 - 1;
      if ((unsigned)ny < (unsigned)H && (unsigned)nx < (unsigned)W)
        __builtin_prefetch(xbase + ((size_t)ny * W + nx) * CCH, 0, 1);
    }

    const _Float16* bb = bp + ((size_t)(kt * 16 + nt0) * 32 + lane) * 16;
#pragma unroll
    for (int t = 0; t < 4; ++t) {
      v16h bf = *(const v16h*)(bb + (size_t)t * 512);
      acc[t] = __builtin_amdgcn_wmma_f32_16x16x32_f16(false, a, false, bf,
                                                      (short)0, acc[t], false, false);
    }
  }
  const int ncol = lane & 15;
  const int mr   = mtile * 16 + (hi << 3);
#pragma unroll
  for (int t = 0; t < 4; ++t) {
    const int n   = nblk * 64 + t * 16 + ncol;
    const float bs = bias[n];
#pragma unroll
    for (int r = 0; r < 8; ++r) {
      float v = acc[t][r] + bs;
      v = v > 0.f ? v : 0.f;
      feat[(size_t)(mr + r) * CCH + n] = (_Float16)v;
    }
  }
}

// ---------------------------------------------------------------------------
// 1x1 head GEMM: feat (M x 256 f16)  x  Bc (256 x 16 f16) -> raw (M x 16 f32).
__global__ __launch_bounds__(128)
void head_wmma(const _Float16* __restrict__ feat, const _Float16* __restrict__ bw,
               const float* __restrict__ bbox_b, const float* __restrict__ conf_b,
               float* __restrict__ raw, int mtiles) {
  const int lane  = threadIdx.x & 31;
  const int mtile = blockIdx.x * 4 + (threadIdx.x >> 5);
  if (mtile >= mtiles) return;  // wave-uniform
  const int hi = lane >> 4;
  const int m  = mtile * 16 + (lane & 15);
  v8f acc = {};
#pragma unroll
  for (int kt = 0; kt < 8; ++kt) {
    const _Float16* s = feat + (size_t)m * CCH + kt * 32 + (hi << 4);
    v16h a = *(const v16h*)(s);
    v16h bf = *(const v16h*)(bw + ((size_t)kt * 32 + lane) * 16);
    acc = __builtin_amdgcn_wmma_f32_16x16x32_f16(false, a, false, bf,
                                                 (short)0, acc, false, false);
  }
  const int n = lane & 15;
  const float bs = (n < 12) ? bbox_b[n] : ((n < 15) ? conf_b[n - 12] : 0.f);
  const int mr = mtile * 16 + (hi << 3);
#pragma unroll
  for (int r = 0; r < 8; ++r)
    raw[(size_t)(mr + r) * 16 + n] = acc[r] + bs;
}

// ---------------------------------------------------------------------------
// Decode: sigmoid(conf), exp-decode boxes with anchor centers, clip to [0,256].
__global__ __launch_bounds__(256)
void decode_kernel(const float* __restrict__ raw0, const float* __restrict__ raw1,
                   float* __restrict__ conf_all, float* __restrict__ boxes_all) {
  const int gid = blockIdx.x * 256 + threadIdx.x;   // BATCH*ATOT = 61440
  const int b = gid / ATOT;
  const int A = gid % ATOT;
  const float* raw;
  int p, j, Hh, s;
  if (A < A0N) {
    p = A / 3; j = A % 3; Hh = 64; s = 4;
    raw = raw0 + ((size_t)(b * 4096 + p)) * 16;
  } else {
    const int A2 = A - A0N;
    p = A2 / 3; j = A2 % 3; Hh = 32; s = 8;
    raw = raw1 + ((size_t)(b * 1024 + p)) * 16;
  }
  const int y = p / Hh, xx = p % Hh;
  const float b0 = raw[4 * j], b1 = raw[4 * j + 1];
  const float b2 = raw[4 * j + 2], b3 = raw[4 * j + 3];
  float cf = raw[12 + j];
  cf = 1.f / (1.f + expf(-cf));
  const float c0 = b0 * 256.f + (float)(xx * s);          // centers = (x*s, y*s)
  const float c1 = b1 * 256.f + (float)(y * s);
  const float h0 = (expf(b2) * 256.f + (float)s) * 0.5f;  // sizes = (s, s)
  const float h1 = (expf(b3) * 256.f + (float)s) * 0.5f;
  float* bo = boxes_all + (size_t)gid * 4;
  bo[0] = clamp256(c0 - h0);
  bo[1] = clamp256(c1 - h1);
  bo[2] = clamp256(c0 + h0);
  bo[3] = clamp256(c1 + h1);
  conf_all[gid] = cf;
}

// ---------------------------------------------------------------------------
// Per-batch descending bitonic sort of packed (conf_bits<<32 | ~idx) keys in
// LDS; then emit top-K conf + gathered boxes. One 1024-thread block per batch.
__global__ __launch_bounds__(1024)
void sort_topk(const float* __restrict__ conf_all, const float* __restrict__ boxes_all,
               float* __restrict__ conf_top, float* __restrict__ boxes_top) {
  extern __shared__ unsigned long long keys[];   // NSORT * 8 = 128 KB
  const int b = blockIdx.x, tid = threadIdx.x;
  for (int i = tid; i < NSORT; i += 1024) {
    unsigned long long k = 0ull;
    if (i < ATOT) {
      const unsigned kb = __float_as_uint(conf_all[(size_t)b * ATOT + i]);
      k = ((unsigned long long)kb << 32) |
          (unsigned long long)(0xFFFFFFFFu - (unsigned)i);
    }
    keys[i] = k;
  }
  __syncthreads();
  for (int kk = 2; kk <= NSORT; kk <<= 1) {
    for (int jj = kk >> 1; jj > 0; jj >>= 1) {
      for (int i = tid; i < NSORT; i += 1024) {
        const int ixj = i ^ jj;
        if (ixj > i) {
          const bool up = ((i & kk) == 0);
          const unsigned long long va = keys[i], vb = keys[ixj];
          if ((va < vb) == up) { keys[i] = vb; keys[ixj] = va; }  // descending
        }
      }
      __syncthreads();
    }
  }
  for (int t = tid; t < KTOP; t += 1024) {
    const unsigned long long k = keys[t];
    conf_top[(size_t)b * KTOP + t] = __uint_as_float((unsigned)(k >> 32));
    const unsigned A = 0xFFFFFFFFu - (unsigned)(k & 0xFFFFFFFFull);
    const float* src = boxes_all + ((size_t)b * ATOT + A) * 4;
    float* dst = boxes_top + ((size_t)b * KTOP + t) * 4;
    dst[0] = src[0]; dst[1] = src[1]; dst[2] = src[2]; dst[3] = src[3];
  }
}

// ---------------------------------------------------------------------------
// Suppression bit-matrix: sup[b][i][w] bit j <=> iou(i, w*32+j) > 0.5.
__global__ __launch_bounds__(256)
void supmat(const float* __restrict__ boxes_top, unsigned* __restrict__ sup) {
  extern __shared__ float bx[];  // KTOP*4 floats = 72 KB
  const int b = blockIdx.y;
  const int i = blockIdx.x * 256 + threadIdx.x;
  const float* src = boxes_top + (size_t)b * KTOP * 4;
  for (int t = threadIdx.x; t < KTOP * 4; t += 256) bx[t] = src[t];
  __syncthreads();
  const float y1 = bx[i * 4], x1 = bx[i * 4 + 1];
  const float y2 = bx[i * 4 + 2], x2 = bx[i * 4 + 3];
  const float ar = (y2 - y1) * (x2 - x1);
  unsigned* row = sup + ((size_t)b * KTOP + i) * SUPW;
  for (int w = 0; w < SUPW; ++w) {
    unsigned bits = 0u;
#pragma unroll 8
    for (int jj = 0; jj < 32; ++jj) {
      const int j = w * 32 + jj;
      const float yy1 = bx[j * 4], xx1 = bx[j * 4 + 1];
      const float yy2 = bx[j * 4 + 2], xx2 = bx[j * 4 + 3];
      const float a2 = (yy2 - yy1) * (xx2 - xx1);
      const float iyv = fminf(y2, yy2) - fmaxf(y1, yy1);
      const float ixv = fminf(x2, xx2) - fmaxf(x1, xx1);
      const float inter = fmaxf(iyv, 0.f) * fmaxf(ixv, 0.f);
      const float uni = fmaxf(ar + a2 - inter, 1e-8f);
      if (inter > 0.5f * uni) bits |= (1u << jj);
    }
    row[w] = bits;
  }
}

// ---------------------------------------------------------------------------
// Single-wave greedy NMS per batch + reference padding logic + final gather.
__global__ __launch_bounds__(32)
void nms_out(const unsigned* __restrict__ sup, const float* __restrict__ conf_top,
             const float* __restrict__ boxes_top, float* __restrict__ out) {
  const int b = blockIdx.x, lane = threadIdx.x;
  __shared__ int kept[PKEEP];
  unsigned rm0 = 0, rm1 = 0, rm2 = 0, rm3 = 0, rm4 = 0;  // 5*32 >= 144 words
  int nk = 0, last = 0;
  for (int i = 0; i < KTOP; ++i) {
    const int w = i >> 5;
    const int c = w >> 5, srcl = w & 31;
    unsigned word;
    switch (c) {
      case 0:  word = __shfl(rm0, srcl); break;
      case 1:  word = __shfl(rm1, srcl); break;
      case 2:  word = __shfl(rm2, srcl); break;
      case 3:  word = __shfl(rm3, srcl); break;
      default: word = __shfl(rm4, srcl); break;
    }
    if (!(word & (1u << (i & 31)))) {           // keep i (uniform branch)
      if (nk < PKEEP) { if (lane == 0) kept[nk] = i; last = i; }
      nk++;
      const unsigned* row = sup + ((size_t)b * KTOP + i) * SUPW;
      rm0 |= row[lane];
      rm1 |= row[32 + lane];
      rm2 |= row[64 + lane];
      rm3 |= row[96 + lane];
      if (128 + lane < SUPW) rm4 |= row[128 + lane];
    }
  }
  __syncthreads();
  const int n = nk < PKEEP ? nk : PKEEP;
  const int add = PKEEP - n;
  int start = KTOP - add;
  if (last + 1 < start) start = last + 1;
  float* conf_out  = out;                         // (4,1000)
  float* boxes_out = out + (size_t)BATCH * PKEEP; // (4,1000,4)
  for (int t = lane; t < PKEEP; t += 32) {
    const int idx = (t < n) ? kept[t] : (start + t - n);
    conf_out[(size_t)b * PKEEP + t] = conf_top[(size_t)b * KTOP + idx];
    const float* s = boxes_top + ((size_t)b * KTOP + idx) * 4;
    float* d = boxes_out + ((size_t)(b * PKEEP + t)) * 4;
    d[0] = s[0]; d[1] = s[1]; d[2] = s[2]; d[3] = s[3];
  }
}

// ---------------------------------------------------------------------------
extern "C" void kernel_launch(void* const* d_in, const int* in_sizes, int n_in,
                              void* d_out, int out_size, void* d_ws, size_t ws_size,
                              hipStream_t stream) {
  (void)in_sizes; (void)n_in; (void)out_size; (void)ws_size;
  const float* x0      = (const float*)d_in[0];
  const float* x1      = (const float*)d_in[1];
  const float* in_w0   = (const float*)d_in[2];
  const float* bbox_w0 = (const float*)d_in[3];
  const float* conf_w0 = (const float*)d_in[4];
  const float* in_w1   = (const float*)d_in[5];
  const float* bbox_w1 = (const float*)d_in[6];
  const float* conf_w1 = (const float*)d_in[7];
  const float* in_b0   = (const float*)d_in[8];
  const float* bbox_b0 = (const float*)d_in[9];
  const float* conf_b0 = (const float*)d_in[10];
  const float* in_b1   = (const float*)d_in[11];
  const float* bbox_b1 = (const float*)d_in[12];
  const float* conf_b1 = (const float*)d_in[13];

  char* ws = (char*)d_ws;
  size_t off = 0;
  auto alloc = [&](size_t bytes) -> void* {
    void* pp = ws + off;
    off = (off + bytes + 255) & ~(size_t)255;
    return pp;
  };
  const size_t BPACK3 = (size_t)KT3 * 16 * 32 * 16;       // halves
  _Float16* bp0   = (_Float16*)alloc(BPACK3 * 2);
  _Float16* bp1   = (_Float16*)alloc(BPACK3 * 2);
  _Float16* bc0   = (_Float16*)alloc((size_t)8 * 32 * 16 * 2);
  _Float16* bc1   = (_Float16*)alloc((size_t)8 * 32 * 16 * 2);
  _Float16* xh0   = (_Float16*)alloc((size_t)16384 * CCH * 2);
  _Float16* xh1   = (_Float16*)alloc((size_t)4096 * CCH * 2);
  _Float16* feat0 = (_Float16*)alloc((size_t)16384 * CCH * 2);
  _Float16* feat1 = (_Float16*)alloc((size_t)4096 * CCH * 2);
  float* raw0      = (float*)alloc((size_t)16384 * 16 * 4);
  float* raw1      = (float*)alloc((size_t)4096 * 16 * 4);
  float* conf_all  = (float*)alloc((size_t)BATCH * ATOT * 4);
  float* boxes_all = (float*)alloc((size_t)BATCH * ATOT * 16);
  float* conf_top  = (float*)alloc((size_t)BATCH * KTOP * 4);
  float* boxes_top = (float*)alloc((size_t)BATCH * KTOP * 16);
  unsigned* sup    = (unsigned*)alloc((size_t)BATCH * KTOP * SUPW * 4);

  // 1) activation f32->f16 + weight packing
  cvt_f16<<<(16384 * CCH) / (256 * 4), 256, 0, stream>>>(x0, xh0);
  cvt_f16<<<(4096 * CCH) / (256 * 4), 256, 0, stream>>>(x1, xh1);
  pack_w3<<<(int)(BPACK3 / 256), 256, 0, stream>>>(in_w0, bp0);
  pack_w3<<<(int)(BPACK3 / 256), 256, 0, stream>>>(in_w1, bp1);
  pack_whead<<<16, 256, 0, stream>>>(bbox_w0, conf_w0, bc0);
  pack_whead<<<16, 256, 0, stream>>>(bbox_w1, conf_w1, bc1);

  // 2) 3x3 conv + ReLU (WMMA implicit GEMM, f16 inputs, f32 accum)
  conv3_wmma<64><<<1024, 128, 0, stream>>>(xh0, in_b0, bp0, feat0);
  conv3_wmma<32><<<256, 128, 0, stream>>>(xh1, in_b1, bp1, feat1);

  // 3) 1x1 heads (WMMA)
  head_wmma<<<256, 128, 0, stream>>>(feat0, bc0, bbox_b0, conf_b0, raw0, 1024);
  head_wmma<<<64, 128, 0, stream>>>(feat1, bc1, bbox_b1, conf_b1, raw1, 256);

  // 4) sigmoid + box decode
  decode_kernel<<<(BATCH * ATOT) / 256, 256, 0, stream>>>(raw0, raw1, conf_all, boxes_all);

  // 5) per-batch top-K via bitonic sort in LDS (128 KB dynamic)
  sort_topk<<<BATCH, 1024, NSORT * sizeof(unsigned long long), stream>>>(
      conf_all, boxes_all, conf_top, boxes_top);

  // 6) suppression bit-matrix (boxes staged in 72 KB LDS)
  supmat<<<dim3(KTOP / 256, BATCH), 256, KTOP * 4 * sizeof(float), stream>>>(boxes_top, sup);

  // 7) greedy NMS + padded gather to output
  nms_out<<<BATCH, 32, 0, stream>>>(sup, conf_top, boxes_top, (float*)d_out);
}
